// MultiHeadSelfAttention_5282809774617
// MI455X (gfx1250) — matrix operand order repaired and checked by reading, not compiled, not run on hardware
//
#include <hip/hip_runtime.h>
#include <hip/hip_bf16.h>

// ---------------------------------------------------------------------------
// MI455X (gfx1250) multi-head self-attention, bf16 WMMA pipeline.
// GEMM tiles staged by the Tensor Data Mover (TENSOR_LOAD_TO_LDS, TENSORcnt);
// attention K/V tiles staged by GLOBAL_LOAD_ASYNC_TO_LDS_B128 (ASYNCcnt).
// ---------------------------------------------------------------------------

typedef __attribute__((ext_vector_type(16))) __bf16 v16bf;
typedef __attribute__((ext_vector_type(8)))  __bf16 v8bf;
typedef __attribute__((ext_vector_type(8)))  float  v8f;
typedef __attribute__((ext_vector_type(4)))  unsigned int v4u;
typedef __attribute__((ext_vector_type(8)))  int v8i;
typedef __attribute__((ext_vector_type(4)))  int v4i;

union BF16x16 { v16bf v; v8bf h[2]; };

__device__ __forceinline__ unsigned short f2bf(float f) {
  unsigned int u = __builtin_bit_cast(unsigned int, f);
  u += 0x7FFFu + ((u >> 16) & 1u);           // round-to-nearest-even
  return (unsigned short)(u >> 16);
}

__device__ __forceinline__ v8f v8f_zero() {
  v8f z;
#pragma unroll
  for (int e = 0; e < 8; e++) z[e] = 0.0f;
  return z;
}

// per-lane 16B async DMA: global -> LDS, tracked by ASYNCcnt
__device__ __forceinline__ void async_b128(unsigned lds_off, const void* g) {
  asm volatile("global_load_async_to_lds_b128 %0, %1, off"
               :: "v"(lds_off), "v"((unsigned long long)g) : "memory");
}

// ---------------------------------------------------------------------------
// TDM: load a 2D bf16 tile [tile_d1 rows x tile_d0 elems] from a row-major
// tensor (row stride = stride_d0 elems) into contiguous LDS at lds_addr.
// D# packing per CDNA5 ISA 8.3/8.4: group0 = {count=1, lds_addr,
// global_addr[56:0], type=2}; group1 = {data_size=2B, tensor dims, tile dims,
// dim0 stride}; higher groups zero (tile_dim2=0 -> 2D).
// Toolchain is the 6-arg clang-23 form: (v4u, v8i, v4i, v4i, v8i, i32).
// ---------------------------------------------------------------------------
__device__ __forceinline__ void tdm_load_2d_bf16(
    unsigned lds_addr, const void* gaddr,
    unsigned tile_d0, unsigned tile_d1,
    unsigned tensor_d0, unsigned tensor_d1, unsigned stride_d0) {
  unsigned long long ga = (unsigned long long)gaddr;
  v4u g0;
  g0[0] = 1u;                                            // count=1 (valid), user
  g0[1] = lds_addr;                                      // LDS byte address
  g0[2] = (unsigned)(ga & 0xFFFFFFFFu);                  // global_addr[31:0]
  g0[3] = (unsigned)((ga >> 32) & 0x01FFFFFFu) | (2u << 30);  // [56:32] | type=2
  v8i g1;
  g1[0] = (int)(1u << 16);                               // data_size=1 (2 bytes)
  g1[1] = (int)((tensor_d0 & 0xFFFFu) << 16);            // tensor_dim0[15:0]
  g1[2] = (int)(((tensor_d0 >> 16) & 0xFFFFu) | ((tensor_d1 & 0xFFFFu) << 16));
  g1[3] = (int)(((tensor_d1 >> 16) & 0xFFFFu) | ((tile_d0 & 0xFFFFu) << 16));
  g1[4] = (int)(tile_d1 & 0xFFFFu);                      // tile_dim1; tile_dim2=0
  g1[5] = (int)stride_d0;                                // tensor_dim0_stride
  g1[6] = 0;
  g1[7] = 0;
  v4i z4; z4[0] = 0; z4[1] = 0; z4[2] = 0; z4[3] = 0;
  v8i z8;
#pragma unroll
  for (int e = 0; e < 8; e++) z8[e] = 0;
  __builtin_amdgcn_tensor_load_to_lds(g0, g1, z4, z4, z8, 0);
}

// ---------------------------------------------------------------------------
// fp32 -> bf16 bulk convert (x activations)
// ---------------------------------------------------------------------------
__global__ __launch_bounds__(256) void cvt_x_kernel(const float* __restrict__ src,
                                                    unsigned short* __restrict__ dst,
                                                    int n) {
  int i = (blockIdx.x * 256 + threadIdx.x) * 8;
  if (i >= n) return;
  float4 a = *(const float4*)(src + i);
  float4 c = *(const float4*)(src + i + 4);
  union { unsigned short s[8]; uint4 u; } o;
  o.s[0] = f2bf(a.x); o.s[1] = f2bf(a.y); o.s[2] = f2bf(a.z); o.s[3] = f2bf(a.w);
  o.s[4] = f2bf(c.x); o.s[5] = f2bf(c.y); o.s[6] = f2bf(c.z); o.s[7] = f2bf(c.w);
  *(uint4*)(dst + i) = o.u;
}

// ---------------------------------------------------------------------------
// fp32 [K,Nw] -> bf16 transposed [Nw,K]  (K fixed at 1024; thread = 4 k-elems)
// ---------------------------------------------------------------------------
__global__ __launch_bounds__(256) void cvt_wT_kernel(const float* __restrict__ w,
                                                     unsigned short* __restrict__ wT,
                                                     int K, int Nw) {
  int gid = blockIdx.x * 256 + threadIdx.x;
  int ncol = gid >> 8;
  int k0 = (gid & 255) * 4;
  if (ncol >= Nw) return;
  union { unsigned short s[4]; uint2 u; } o;
#pragma unroll
  for (int i = 0; i < 4; i++) o.s[i] = f2bf(w[(size_t)(k0 + i) * Nw + ncol]);
  *(uint2*)(wT + (size_t)ncol * K + k0) = o.u;
}

// ---------------------------------------------------------------------------
// Tiled bf16 GEMM: C[M,N] = A[M,K] * Bt[N,K]^T (+bias)
// Block = 256 thr (8 waves), tile 128x256, K-step 32; wave = 64x64
// -> 16 WMMAs per K-step per wave. Tiles staged by TDM (wave 0 issues two
// TENSOR_LOAD_TO_LDS per tile; in-order per wave), double buffered.
// mode 0: QKV epilogue (bias, Q*0.125->[B,H,N,D], K->[B,H,N,D],
//                       V->[B,H,D,N] transposed), bf16 outputs.
// mode 1: proj epilogue, fp32 out = val + bias.
// ---------------------------------------------------------------------------
__global__ __launch_bounds__(256) void gemm_bf16_kernel(
    const unsigned short* __restrict__ A, const unsigned short* __restrict__ Bt,
    int M, int N, int K, int mode, const float* __restrict__ bias,
    float* __restrict__ outF, unsigned short* __restrict__ qbuf,
    unsigned short* __restrict__ kbuf, unsigned short* __restrict__ vbuf) {
  __shared__ unsigned short Atile[2 * 128 * 32];   // [buf][m][k]  2x8KB
  __shared__ unsigned short Btile[2 * 256 * 32];   // [buf][n][k]  2x16KB

  const int tid = threadIdx.x;
  const int lane = tid & 31;
  const int wave = tid >> 5;
  const int waveM = wave >> 2;   // 0..1
  const int waveN = wave & 3;    // 0..3
  const int m16 = lane & 15;
  const int hi = lane >> 4;
  const int kb = hi * 8;         // A-operand K base (ISA 16-bit A 16x32 layout)

  const int n0 = blockIdx.x * 256;
  const int m0 = blockIdx.y * 128;

  const unsigned ldsA = (unsigned)(unsigned long long)(void*)Atile;
  const unsigned ldsB = (unsigned)(unsigned long long)(void*)Btile;

  v8f acc[4][4];
#pragma unroll
  for (int i = 0; i < 4; i++)
#pragma unroll
    for (int j = 0; j < 4; j++) acc[i][j] = v8f_zero();

  const int nT = K / 32;

  if (wave == 0) {   // TDM is wave-level; one issuing wave per block
    tdm_load_2d_bf16(ldsA, A + (size_t)m0 * K, 32, 128, K, M, K);
    tdm_load_2d_bf16(ldsB, Bt + (size_t)n0 * K, 32, 256, K, N, K);
  }

  for (int t = 0; t < nT; ++t) {
    if (wave == 0) {
      if (t + 1 < nT) {
        const int k0n = (t + 1) * 32;
        const unsigned nxt = (unsigned)((t + 1) & 1);
        tdm_load_2d_bf16(ldsA + nxt * 8192u, A + (size_t)m0 * K + k0n, 32, 128, K, M, K);
        tdm_load_2d_bf16(ldsB + nxt * 16384u, Bt + (size_t)n0 * K + k0n, 32, 256, K, N, K);
        __builtin_amdgcn_s_wait_tensorcnt(2);   // tile t landed (in-order)
      } else {
        __builtin_amdgcn_s_wait_tensorcnt(0);
      }
    }
    __syncthreads();

    const int cur = t & 1;
    BF16x16 af[4], bfg[4];
#pragma unroll
    for (int i = 0; i < 4; i++) {
      int base = cur * 4096 + (waveM * 64 + i * 16 + m16) * 32;
      af[i].h[0] = *(const v8bf*)&Atile[base + kb];
      af[i].h[1] = *(const v8bf*)&Atile[base + kb + 16];
    }
#pragma unroll
    for (int j = 0; j < 4; j++) {
      int base = cur * 8192 + (waveN * 64 + j * 16 + m16) * 32;
      bfg[j].h[0] = *(const v8bf*)&Btile[base + kb];
      bfg[j].h[1] = *(const v8bf*)&Btile[base + kb + 16];
    }
#pragma unroll
    for (int i = 0; i < 4; i++)
#pragma unroll
      for (int j = 0; j < 4; j++)
        acc[i][j] = __builtin_amdgcn_wmma_f32_16x16x32_bf16(
            false, af[i].v, false, bfg[j].v, (short)0, acc[i][j], false, false);

    __syncthreads();   // readers done before buf reuse next iteration
  }

  // epilogue
#pragma unroll
  for (int i = 0; i < 4; i++) {
#pragma unroll
    for (int j = 0; j < 4; j++) {
      int gn = n0 + waveN * 64 + j * 16 + m16;
      float bv = bias[gn];
#pragma unroll
      for (int r = 0; r < 8; r++) {
        int gm = m0 + waveM * 64 + i * 16 + r + hi * 8;
        float val = acc[i][j][r] + bv;
        if (mode == 1) {
          outF[(size_t)gm * N + gn] = val;
        } else {
          int b = gm >> 11, n = gm & 2047;
          int sec = gn >> 10, c = gn & 1023;
          int h = c >> 6, d = c & 63;
          int bh = b * 16 + h;
          if (sec == 0)       qbuf[((size_t)(bh * 2048 + n)) * 64 + d] = f2bf(val * 0.125f);
          else if (sec == 1)  kbuf[((size_t)(bh * 2048 + n)) * 64 + d] = f2bf(val);
          else                vbuf[((size_t)(bh * 64 + d)) * 2048 + n] = f2bf(val);
        }
      }
    }
  }
}

// ---------------------------------------------------------------------------
// Flash attention: block = 8 waves x 16 q-rows = 128 q rows per (b,h) slice.
// kv tiles of 32 staged via async-to-LDS (1 K + 1 V async per wave per tile),
// double buffered. S = Q K^T (4 WMMAs), online softmax via half-wave
// butterflies, P -> wave-private LDS -> A-fragment, O += P V (4 WMMAs).
// grid = B*H * (2048/128) = 1024 blocks.
// ---------------------------------------------------------------------------
__global__ __launch_bounds__(256) void attn_kernel(
    const unsigned short* __restrict__ qbuf,   // [B,H,N,D] bf16, pre-scaled 1/8
    const unsigned short* __restrict__ kbuf,   // [B,H,N,D] bf16
    const unsigned short* __restrict__ vbuf,   // [B,H,D,N] bf16 (transposed)
    unsigned short* __restrict__ attnbuf) {    // [B,N,H*D] bf16
  __shared__ unsigned short Ktile[2 * 32 * 64];   // [buf][kv][d]
  __shared__ unsigned short Vtile[2 * 64 * 32];   // [buf][d][kv]
  __shared__ unsigned short Ptile[8 * 16 * 32];

  const int tid = threadIdx.x;
  const int lane = tid & 31;
  const int wave = tid >> 5;
  const int m16 = lane & 15;
  const int hi = lane >> 4;
  const int kb = hi * 8;

  const int qb = blockIdx.x & 15;
  const int bh = blockIdx.x >> 4;
  const int b = bh >> 4;
  const int h = bh & 15;
  const int qrow0 = qb * 128 + wave * 16;

  const unsigned ldsK = (unsigned)(unsigned long long)(void*)Ktile;
  const unsigned ldsV = (unsigned)(unsigned long long)(void*)Vtile;
  const int krow = wave * 4 + (lane >> 3), kdp = (lane & 7) * 8;
  const int vrow = wave * 8 + (lane >> 2), vkp = (lane & 3) * 8;

  // Q fragments (A-layout), rows m16, d split 0..31 / 32..63
  BF16x16 q0, q1;
  {
    const unsigned short* qp = qbuf + ((size_t)bh * 2048 + qrow0 + m16) * 64;
    q0.h[0] = *(const v8bf*)(qp + kb);
    q0.h[1] = *(const v8bf*)(qp + kb + 16);
    q1.h[0] = *(const v8bf*)(qp + 32 + kb);
    q1.h[1] = *(const v8bf*)(qp + 32 + kb + 16);
  }

  v8f o[4];
#pragma unroll
  for (int j = 0; j < 4; j++) o[j] = v8f_zero();
  float mr[8], lr[8];
#pragma unroll
  for (int r = 0; r < 8; r++) { mr[r] = -__builtin_inff(); lr[r] = 0.0f; }

  unsigned short* Pw = &Ptile[wave * 16 * 32];

  // issue kv tile 0 -> buf 0
  async_b128(ldsK + (unsigned)(krow * 64 + kdp) * 2u,
             kbuf + ((size_t)bh * 2048 + krow) * 64 + kdp);
  async_b128(ldsV + (unsigned)(vrow * 32 + vkp) * 2u,
             vbuf + ((size_t)bh * 64 + vrow) * 2048 + vkp);

  for (int t = 0; t < 64; ++t) {
    if (t + 1 < 64) {
      const int kv0n = (t + 1) * 32;
      const int nxt = (t + 1) & 1;
      async_b128(ldsK + (unsigned)(nxt * 2048 + krow * 64 + kdp) * 2u,
                 kbuf + ((size_t)bh * 2048 + kv0n + krow) * 64 + kdp);
      async_b128(ldsV + (unsigned)(nxt * 2048 + vrow * 32 + vkp) * 2u,
                 vbuf + ((size_t)bh * 64 + vrow) * 2048 + kv0n + vkp);
      asm volatile("s_wait_asynccnt 0x2" ::: "memory");
    } else {
      asm volatile("s_wait_asynccnt 0x0" ::: "memory");
    }
    __syncthreads();
    const int cur = (t & 1) * 2048;

    // S = Q K^T for 2 kv sub-tiles of 16
    v8f s[2];
#pragma unroll
    for (int nn = 0; nn < 2; nn++) {
      int row = cur + (nn * 16 + m16) * 64;
      BF16x16 k0f, k1f;
      k0f.h[0] = *(const v8bf*)&Ktile[row + kb];
      k0f.h[1] = *(const v8bf*)&Ktile[row + kb + 16];
      k1f.h[0] = *(const v8bf*)&Ktile[row + 32 + kb];
      k1f.h[1] = *(const v8bf*)&Ktile[row + 32 + kb + 16];
      v8f sz = v8f_zero();
      sz = __builtin_amdgcn_wmma_f32_16x16x32_bf16(false, q0.v, false, k0f.v,
                                                   (short)0, sz, false, false);
      sz = __builtin_amdgcn_wmma_f32_16x16x32_bf16(false, q1.v, false, k1f.v,
                                                   (short)0, sz, false, false);
      s[nn] = sz;
    }

    // online softmax (row fixed per half-wave; butterfly over 16 lanes)
    float p0a[8], p1a[8], alpha[8];
#pragma unroll
    for (int r = 0; r < 8; r++) {
      float tmax = fmaxf(s[0][r], s[1][r]);
      tmax = fmaxf(tmax, __shfl_xor(tmax, 1));
      tmax = fmaxf(tmax, __shfl_xor(tmax, 2));
      tmax = fmaxf(tmax, __shfl_xor(tmax, 4));
      tmax = fmaxf(tmax, __shfl_xor(tmax, 8));
      float nm = fmaxf(mr[r], tmax);
      alpha[r] = __expf(mr[r] - nm);
      mr[r] = nm;
      float p0 = __expf(s[0][r] - nm);
      float p1 = __expf(s[1][r] - nm);
      p0a[r] = p0; p1a[r] = p1;
      float rs = p0 + p1;
      rs += __shfl_xor(rs, 1);
      rs += __shfl_xor(rs, 2);
      rs += __shfl_xor(rs, 4);
      rs += __shfl_xor(rs, 8);
      lr[r] = lr[r] * alpha[r] + rs;
    }
#pragma unroll
    for (int j = 0; j < 4; j++)
#pragma unroll
      for (int r = 0; r < 8; r++) o[j][r] *= alpha[r];

    // P (bf16) to wave-private LDS, then reload in A-operand layout
#pragma unroll
    for (int r = 0; r < 8; r++) {
      int q = r + hi * 8;
      Pw[q * 32 + m16] = f2bf(p0a[r]);
      Pw[q * 32 + 16 + m16] = f2bf(p1a[r]);
    }
    asm volatile("s_wait_dscnt 0x0" ::: "memory");   // wave-local round trip
    BF16x16 pf;
    pf.h[0] = *(const v8bf*)&Pw[m16 * 32 + kb];
    pf.h[1] = *(const v8bf*)&Pw[m16 * 32 + kb + 16];

    // O += P V (Vt in LDS: B-operand rows are d, contiguous kv)
#pragma unroll
    for (int j = 0; j < 4; j++) {
      int row = cur + (j * 16 + m16) * 32;
      BF16x16 vf;
      vf.h[0] = *(const v8bf*)&Vtile[row + kb];
      vf.h[1] = *(const v8bf*)&Vtile[row + kb + 16];
      o[j] = __builtin_amdgcn_wmma_f32_16x16x32_bf16(false, pf.v, false, vf.v,
                                                     (short)0, o[j], false, false);
    }
    __syncthreads();   // tile readers done before async reuse of this buffer
  }

  // normalize + store [B,N,H*D] bf16
#pragma unroll
  for (int r = 0; r < 8; r++) {
    float inv = 1.0f / lr[r];
    int gq = qrow0 + r + hi * 8;
    size_t base = ((size_t)b * 2048 + gq) * 1024 + h * 64;
#pragma unroll
    for (int j = 0; j < 4; j++)
      attnbuf[base + j * 16 + m16] = f2bf(o[j][r] * inv);
  }
}

// ---------------------------------------------------------------------------
extern "C" void kernel_launch(void* const* d_in, const int* in_sizes, int n_in,
                              void* d_out, int out_size, void* d_ws, size_t ws_size,
                              hipStream_t stream) {
  const float* x      = (const float*)d_in[0];
  const float* w_qkv  = (const float*)d_in[1];
  const float* b_qkv  = (const float*)d_in[2];
  const float* w_proj = (const float*)d_in[3];
  const float* b_proj = (const float*)d_in[4];
  float* out = (float*)d_out;

  char* ws = (char*)d_ws;
  unsigned short* xbf    = (unsigned short*)(ws);                       // 16 MB
  unsigned short* wqkvT  = (unsigned short*)(ws + 16777216);            // 6 MB
  unsigned short* wprojT = (unsigned short*)(ws + 23068672);            // 2 MB
  unsigned short* qbuf   = (unsigned short*)(ws + 25165824);            // 16 MB
  unsigned short* kbuf   = (unsigned short*)(ws + 41943040);            // 16 MB
  unsigned short* vbuf   = (unsigned short*)(ws + 58720256);            // 16 MB
  unsigned short* abuf   = (unsigned short*)(ws + 75497472);            // 16 MB

  cvt_x_kernel<<<4096, 256, 0, stream>>>(x, xbf, 8388608);
  cvt_wT_kernel<<<3072, 256, 0, stream>>>(w_qkv, wqkvT, 1024, 3072);
  cvt_wT_kernel<<<1024, 256, 0, stream>>>(w_proj, wprojT, 1024, 1024);

  // QKV: [8192,1024] x [1024,3072], block tile 128x256
  gemm_bf16_kernel<<<dim3(12, 64), 256, 0, stream>>>(
      xbf, wqkvT, 8192, 3072, 1024, 0, b_qkv, nullptr, qbuf, kbuf, vbuf);

  attn_kernel<<<1024, 256, 0, stream>>>(qbuf, kbuf, vbuf, abuf);

  // proj: [8192,1024] x [1024,1024] -> fp32 out
  gemm_bf16_kernel<<<dim3(4, 64), 256, 0, stream>>>(
      abuf, wprojT, 8192, 1024, 1024, 1, b_proj, out, nullptr, nullptr, nullptr);
}
